// SSM_8392366097155
// MI455X (gfx1250) — matrix operand order repaired and checked by reading, not compiled, not run on hardware
//
#include <hip/hip_runtime.h>
#include <hip/hip_bf16.h>
#include <math.h>

// ---------------------------------------------------------------------------
// Types for CDNA5 WMMA (wave32): 16x16x32 bf16 -> f32
// ---------------------------------------------------------------------------
typedef __bf16  bf16_t;
typedef __bf16  v16bf __attribute__((ext_vector_type(16)));
typedef __bf16  v4bf  __attribute__((ext_vector_type(4)));
typedef float   v8f   __attribute__((ext_vector_type(8)));

// pointee type the async-LDS builtin expects (per hipcc diagnostic):
// 'int __attribute__((__vector_size__(4 * sizeof(int)))) *'
typedef int v4i_vs __attribute__((vector_size(16)));

#define D_MODEL   1024
#define D_HIDDEN  256
#define BATCH     4
#define SEQ       4096
#define MTOK      (BATCH * SEQ)       // 16384 rows
#define NCHUNK    64                  // scan chunks along T
#define CHLEN     (SEQ / NCHUNK)      // 64 steps per chunk

__device__ __forceinline__ float softplus_f(float v) {
    return (v > 20.0f) ? v : log1pf(expf(v));
}
__device__ __forceinline__ float clamp20(float v) {
    return fminf(20.0f, fmaxf(-20.0f, v));
}

// ---------------------------------------------------------------------------
// CDNA5 async global->LDS copy (ASYNCcnt path), 16 bytes per lane.
// LDS operand: low 32 bits of the flat shared address are the LDS offset
// (ISA 10.2 aperture rules), so an integer-truncate cast is sufficient.
// ---------------------------------------------------------------------------
__device__ __forceinline__ void async_copy16(const void* g, void* l) {
#if __has_builtin(__builtin_amdgcn_global_load_async_to_lds_b128)
    __builtin_amdgcn_global_load_async_to_lds_b128(
        (__attribute__((address_space(1))) v4i_vs*)(unsigned long long)(uintptr_t)g,
        (__attribute__((address_space(3))) v4i_vs*)(unsigned)(uintptr_t)l,
        0, 0);
#else
    asm volatile("global_load_async_to_lds_b128 %0, %1, off"
                 :: "v"((unsigned)(uintptr_t)l),
                    "v"((unsigned long long)(uintptr_t)g)
                 : "memory");
#endif
}

__device__ __forceinline__ void wait_async_le4() {
#if __has_builtin(__builtin_amdgcn_s_wait_asynccnt)
    __builtin_amdgcn_s_wait_asynccnt(4);
#else
    asm volatile("s_wait_asynccnt 4" ::: "memory");
#endif
}
__device__ __forceinline__ void wait_async_le0() {
#if __has_builtin(__builtin_amdgcn_s_wait_asynccnt)
    __builtin_amdgcn_s_wait_asynccnt(0);
#else
    asm volatile("s_wait_asynccnt 0" ::: "memory");
#endif
}

// ---------------------------------------------------------------------------
// 1) casts / transposes / A_log precompute
// ---------------------------------------------------------------------------
__global__ void k_cast_x(const float* __restrict__ x, bf16_t* __restrict__ xb) {
    size_t i = (size_t)blockIdx.x * blockDim.x + threadIdx.x;   // i indexes float4
    const float4 v = ((const float4*)x)[i];
    v4bf o;
    o[0] = (__bf16)v.x; o[1] = (__bf16)v.y; o[2] = (__bf16)v.z; o[3] = (__bf16)v.w;
    ((v4bf*)xb)[i] = o;
}

// W1T[n][k], n in [0,512): n<256 -> Wdelta[:,n] ; n>=256 -> Wb[:,n-256]
__global__ void k_build_w1t(const float* __restrict__ Wdelta,
                            const float* __restrict__ Wb,
                            bf16_t* __restrict__ W1T) {
    int idx = blockIdx.x * blockDim.x + threadIdx.x;            // 512*1024
    int n = idx >> 10, k = idx & 1023;
    float v = (n < D_HIDDEN) ? Wdelta[(size_t)k * D_HIDDEN + n]
                             : Wb[(size_t)k * D_HIDDEN + (n - D_HIDDEN)];
    W1T[idx] = (__bf16)v;
}

__global__ void k_build_wdt(const float* __restrict__ Wd, bf16_t* __restrict__ WdT) {
    int idx = blockIdx.x * blockDim.x + threadIdx.x;            // 1024*1024
    int n = idx >> 10, k = idx & 1023;
    WdT[idx] = (__bf16)Wd[(size_t)k * D_MODEL + n];
}

__global__ void k_build_wct(const float* __restrict__ Wc, bf16_t* __restrict__ WcT) {
    int idx = blockIdx.x * blockDim.x + threadIdx.x;            // 1024*256
    int n = idx >> 8, k = idx & 255;                            // n: out col, k: hidden
    WcT[idx] = (__bf16)Wc[(size_t)k * D_MODEL + n];
}

__global__ void k_alog(const float* __restrict__ A, float* __restrict__ Alog) {
    int h = threadIdx.x;
    if (h < D_HIDDEN) Alog[h] = -softplus_f(A[h]);
}

// ---------------------------------------------------------------------------
// 2) WMMA GEMM: C[M x N] = A[M x K](bf16, row-major) * B[N x K](bf16, "NT")
//    optional second product accumulated in, optional column bias(es).
//    Block tile 128x128, 8 waves, wave tile 32x64 (2x4 fragments).
//    Tiles staged with async global->LDS copies, double-buffered.
// ---------------------------------------------------------------------------
#define BM 128
#define BN 128
#define BK 32
#define BKP 40   // padded LDS row stride (elements) to stagger banks

union FragU { v16bf v; uint4 q[2]; };

__global__ __launch_bounds__(256)
void k_gemm(const bf16_t* __restrict__ A,  const bf16_t* __restrict__ Bw,
            float* __restrict__ C,
            int K,  int lda,  int ldb,  int ldc,
            const bf16_t* __restrict__ A2, const bf16_t* __restrict__ B2,
            int K2, int lda2, int ldb2,
            const float* __restrict__ bias1, const float* __restrict__ bias2)
{
    __shared__ bf16_t Asm[2][BM * BKP];
    __shared__ bf16_t Bsm[2][BN * BKP];

    const int tid  = threadIdx.x;
    const int m0   = blockIdx.x * BM;
    const int n0   = blockIdx.y * BN;
    const int lane = tid & 31;
    const int w    = tid >> 5;
    const int wm   = w >> 1;        // 0..3  -> 32-row strip
    const int wn   = w & 1;         // 0..1  -> 64-col strip
    const int l    = lane & 15;
    const int sub  = lane >> 4;     // half-wave selector

    v8f acc[2][4];
#pragma unroll
    for (int i = 0; i < 2; ++i)
#pragma unroll
        for (int j = 0; j < 4; ++j) acc[i][j] = {};

    int buf = 0;

    // issue the 4 async copies (2 A chunks + 2 B chunks, 16B each) for one tile
    auto stage = [&](const bf16_t* Ap, const bf16_t* Bp, int k0,
                     int ldac, int ldbc, int bsel) {
#pragma unroll
        for (int c = tid; c < (BM * BK) / 8; c += 256) {
            int row = c >> 2, seg = c & 3;
            async_copy16(Ap + (size_t)(m0 + row) * ldac + k0 + seg * 8,
                         &Asm[bsel][row * BKP + seg * 8]);
        }
#pragma unroll
        for (int c = tid; c < (BN * BK) / 8; c += 256) {
            int row = c >> 2, seg = c & 3;
            async_copy16(Bp + (size_t)(n0 + row) * ldbc + k0 + seg * 8,
                         &Bsm[bsel][row * BKP + seg * 8]);
        }
    };

    auto mm = [&](const bf16_t* Ap, const bf16_t* Bp, int Kc, int ldac, int ldbc) {
        stage(Ap, Bp, 0, ldac, ldbc, buf);             // prefetch first tile
        for (int k0 = 0; k0 < Kc; k0 += BK) {
            const bool hasNext = (k0 + BK) < Kc;
            __syncthreads();            // prior compute done before touching buf^1
            if (hasNext) {
                stage(Ap, Bp, k0 + BK, ldac, ldbc, buf ^ 1);
                wait_async_le4();       // 4 older ops (= current tile) landed
            } else {
                wait_async_le0();
            }
            __syncthreads();            // current tile visible to all waves

            // A fragments: 16x32, lane<16: M=l, K elems = sub*8.. / 16+sub*8..
            v16bf afr[2], bfr[4];
#pragma unroll
            for (int fm = 0; fm < 2; ++fm) {
                int row = wm * 32 + fm * 16 + l;
                FragU u;
                u.q[0] = *(const uint4*)&Asm[buf][row * BKP + sub * 8];
                u.q[1] = *(const uint4*)&Asm[buf][row * BKP + 16 + sub * 8];
                afr[fm] = u.v;
            }
            // B fragments: 32x16 (KxN): K elems = sub*8.. / 16+sub*8.. (same as A)
#pragma unroll
            for (int fn = 0; fn < 4; ++fn) {
                int n = wn * 64 + fn * 16 + l;
                FragU u;
                u.q[0] = *(const uint4*)&Bsm[buf][n * BKP + sub * 8];
                u.q[1] = *(const uint4*)&Bsm[buf][n * BKP + 16 + sub * 8];
                bfr[fn] = u.v;
            }
#pragma unroll
            for (int fm = 0; fm < 2; ++fm)
#pragma unroll
                for (int fn = 0; fn < 4; ++fn)
                    acc[fm][fn] = __builtin_amdgcn_wmma_f32_16x16x32_bf16(
                        false, afr[fm], false, bfr[fn],
                        (short)0, acc[fm][fn], false, false);
            buf ^= 1;
        }
    };

    mm(A, Bw, K, lda, ldb);
    if (A2) mm(A2, B2, K2, lda2, ldb2);

    // epilogue: C layout per ISA — VGPR r: lanes0-15 M=r, lanes16-31 M=8+r
#pragma unroll
    for (int fm = 0; fm < 2; ++fm) {
#pragma unroll
        for (int fn = 0; fn < 4; ++fn) {
            int col = n0 + wn * 64 + fn * 16 + l;
            float bsum = 0.0f;
            if (bias1) bsum += bias1[col];
            if (bias2) bsum += bias2[col];
            int mrow = m0 + wm * 32 + fm * 16 + sub * 8;
#pragma unroll
            for (int r = 0; r < 8; ++r)
                C[(size_t)(mrow + r) * ldc + col] = acc[fm][fn][r] + bsum;
        }
    }
}

// ---------------------------------------------------------------------------
// 3) elementwise prep: delta=softplus(G[:,h]+bdelta), a=delta*A_log,
//    Bbar=delta*(G[:,256+h]+bb)
// ---------------------------------------------------------------------------
__global__ void k_prep(const float* __restrict__ G,
                       const float* __restrict__ bdelta,
                       const float* __restrict__ bb,
                       const float* __restrict__ Alog,
                       float* __restrict__ a, float* __restrict__ Bbar)
{
    size_t idx = (size_t)blockIdx.x * blockDim.x + threadIdx.x;  // M*H
    int m = (int)(idx >> 8), h = (int)(idx & 255);
    float gd    = G[(size_t)m * 512 + h] + bdelta[h];
    float delta = softplus_f(gd);
    a[idx]      = delta * Alog[h];
    Bbar[idx]   = delta * (G[(size_t)m * 512 + 256 + h] + bb[h]);
}

// ---------------------------------------------------------------------------
// 4) chunked scan along T (3 phases). block = (b, chunk), thread = h channel.
// ---------------------------------------------------------------------------
__global__ void k_scan_partial(const float* __restrict__ a, float* __restrict__ sumA)
{
    int b = blockIdx.x / NCHUNK, ch = blockIdx.x % NCHUNK, hc = threadIdx.x;
    float s = 0.0f;
    for (int i = 0; i < CHLEN; ++i) {
        int t = ch * CHLEN + i;
        s += a[((size_t)(b * SEQ + t) << 8) + hc];
    }
    sumA[((size_t)(b * NCHUNK + ch) << 8) + hc] = s;
}

// exclusive scan over chunks, in place; one thread per (b,h)
__global__ void k_scan_chunks(float* __restrict__ buf)
{
    int tid = blockIdx.x * blockDim.x + threadIdx.x;   // B*H = 1024
    int b = tid >> 8, hc = tid & 255;
    float run = 0.0f;
    for (int c = 0; c < NCHUNK; ++c) {
        size_t idx = ((size_t)(b * NCHUNK + c) << 8) + hc;
        float v = buf[idx];
        buf[idx] = run;
        run += v;
    }
}

// weighted local cumsum of Bshift; Bbar overwritten with local cumsum L
__global__ void k_scan_bshift(const float* __restrict__ a,
                              float* __restrict__ Bbar,   // in: Bbar, out: L
                              const float* __restrict__ prefA,
                              float* __restrict__ sumD)
{
    int b = blockIdx.x / NCHUNK, ch = blockIdx.x % NCHUNK, hc = threadIdx.x;
    float s = prefA[((size_t)(b * NCHUNK + ch) << 8) + hc];
    float c = 0.0f;
    for (int i = 0; i < CHLEN; ++i) {
        int t = ch * CHLEN + i;
        size_t off = ((size_t)(b * SEQ + t) << 8) + hc;
        float shifted = clamp20(s);                 // A_log_cum at t-1
        c += Bbar[off] * __expf(-shifted);
        Bbar[off] = c;                              // local cumsum
        s += a[off];
    }
    sumD[((size_t)(b * NCHUNK + ch) << 8) + hc] = c;
}

// finalize: h[t] = exp(clamp(S_t)) * (prefD + L[t]) -> bf16 for final GEMM
__global__ void k_scan_final(const float* __restrict__ a,
                             const float* __restrict__ L,
                             const float* __restrict__ prefA,
                             const float* __restrict__ prefD,
                             bf16_t* __restrict__ hb)
{
    int b = blockIdx.x / NCHUNK, ch = blockIdx.x % NCHUNK, hc = threadIdx.x;
    size_t pidx = ((size_t)(b * NCHUNK + ch) << 8) + hc;
    float s = prefA[pidx];
    float cp = prefD[pidx];
    for (int i = 0; i < CHLEN; ++i) {
        int t = ch * CHLEN + i;
        size_t off = ((size_t)(b * SEQ + t) << 8) + hc;
        s += a[off];
        float val = __expf(clamp20(s)) * (cp + L[off]);
        hb[off] = (__bf16)val;
    }
}

// ---------------------------------------------------------------------------
// host launch
// ---------------------------------------------------------------------------
extern "C" void kernel_launch(void* const* d_in, const int* in_sizes, int n_in,
                              void* d_out, int out_size, void* d_ws, size_t ws_size,
                              hipStream_t stream)
{
    (void)in_sizes; (void)n_in; (void)out_size; (void)ws_size;
    const float* x      = (const float*)d_in[0];
    const float* A      = (const float*)d_in[1];
    const float* Wb     = (const float*)d_in[2];
    const float* bb     = (const float*)d_in[3];
    const float* Wc     = (const float*)d_in[4];
    const float* bc     = (const float*)d_in[5];
    const float* Wd     = (const float*)d_in[6];
    const float* bd     = (const float*)d_in[7];
    const float* Wdelta = (const float*)d_in[8];
    const float* bdelta = (const float*)d_in[9];
    float* out = (float*)d_out;

    char* p = (char*)d_ws;
    auto alloc = [&](size_t bytes) -> void* {
        void* r = (void*)p;
        p += (bytes + 255) & ~(size_t)255;
        return r;
    };
    bf16_t* xb   = (bf16_t*)alloc((size_t)MTOK * D_MODEL * 2);       // 32 MB
    bf16_t* W1T  = (bf16_t*)alloc((size_t)512 * 1024 * 2);           // 1 MB
    bf16_t* WdT  = (bf16_t*)alloc((size_t)1024 * 1024 * 2);          // 2 MB
    bf16_t* WcT  = (bf16_t*)alloc((size_t)1024 * 256 * 2);           // 0.5 MB
    float*  Alog = (float*) alloc(256 * 4);
    float*  G    = (float*) alloc((size_t)MTOK * 512 * 4);           // 32 MB
    float*  av   = (float*) alloc((size_t)MTOK * D_HIDDEN * 4);      // 16 MB
    float*  Bbar = (float*) alloc((size_t)MTOK * D_HIDDEN * 4);      // 16 MB (becomes L)
    float*  sumA = (float*) alloc((size_t)BATCH * NCHUNK * 256 * 4);
    float*  sumD = (float*) alloc((size_t)BATCH * NCHUNK * 256 * 4);
    bf16_t* hb   = (bf16_t*)G;   // reuse G (dead after k_prep) for bf16 h

    // 1) casts / transposes / A_log
    k_cast_x  <<<(MTOK * D_MODEL / 4) / 256, 256, 0, stream>>>(x, xb);
    k_build_w1t<<<(512 * 1024) / 256, 256, 0, stream>>>(Wdelta, Wb, W1T);
    k_build_wdt<<<(1024 * 1024) / 256, 256, 0, stream>>>(Wd, WdT);
    k_build_wct<<<(1024 * 256) / 256, 256, 0, stream>>>(Wc, WcT);
    k_alog     <<<1, 256, 0, stream>>>(A, Alog);

    // 2) G = x @ [Wdelta | Wb]   (M x 512, K = 1024)
    k_gemm<<<dim3(MTOK / BM, 512 / BN), 256, 0, stream>>>(
        xb, W1T, G,
        /*K=*/D_MODEL, /*lda=*/D_MODEL, /*ldb=*/D_MODEL, /*ldc=*/512,
        nullptr, nullptr, 0, 0, 0, nullptr, nullptr);

    // 3) a, Bbar
    k_prep<<<((size_t)MTOK * D_HIDDEN) / 256, 256, 0, stream>>>(G, bdelta, bb, Alog, av, Bbar);

    // 4) chunked scan
    k_scan_partial<<<BATCH * NCHUNK, 256, 0, stream>>>(av, sumA);
    k_scan_chunks <<<BATCH, 256, 0, stream>>>(sumA);                 // -> exclusive prefA
    k_scan_bshift <<<BATCH * NCHUNK, 256, 0, stream>>>(av, Bbar, sumA, sumD);
    k_scan_chunks <<<BATCH, 256, 0, stream>>>(sumD);                 // -> exclusive prefD
    k_scan_final  <<<BATCH * NCHUNK, 256, 0, stream>>>(av, Bbar, sumA, sumD, hb);

    // 5) out = x @ Wd + h @ Wc + bc + bd   (M x 1024)
    k_gemm<<<dim3(MTOK / BM, D_MODEL / BN), 256, 0, stream>>>(
        xb, WdT, out,
        /*K=*/D_MODEL, /*lda=*/D_MODEL, /*ldb=*/D_MODEL, /*ldc=*/D_MODEL,
        hb, WcT, /*K2=*/D_HIDDEN, /*lda2=*/D_HIDDEN, /*ldb2=*/D_HIDDEN,
        bc, bd);
}
